// LSTM_6691559047736
// MI455X (gfx1250) — matrix operand order repaired and checked by reading, not compiled, not run on hardware
//
#include <hip/hip_runtime.h>
#include <hip/hip_bf16.h>

typedef __attribute__((ext_vector_type(16))) _Float16 v16h;
typedef __attribute__((ext_vector_type(8)))  float    v8f;

#define T_STEPS 1000
#define BATCH   256
#define D_IN    440
#define HID     20
#define G4      80      // 4*HID
#define KPAD    448     // D_IN padded up to multiple of 32

__device__ __forceinline__ float sigmoid_f(float x) {
    return 1.0f / (1.0f + __expf(-x));
}
__device__ __forceinline__ float tanh_f(float x) {
    return 2.0f / (1.0f + __expf(-2.0f * x)) - 1.0f;
}

// ---------------------------------------------------------------------------
// Kernel 0: convert Wx0 (80 x 440 f32) -> f16, zero-padded to 80 x 448, laid
// out so B-fragment loads in the GEMM are contiguous, 32B-aligned v16h reads.
// ---------------------------------------------------------------------------
__global__ void cvt_w0_f16(const float* __restrict__ Wx0,
                           _Float16* __restrict__ W16) {
    int idx = blockIdx.x * 256 + threadIdx.x;
    if (idx >= G4 * KPAD) return;
    int col = idx / KPAD;
    int k   = idx - col * KPAD;
    W16[idx] = (k < D_IN) ? (_Float16)Wx0[col * D_IN + k] : (_Float16)0.0f;
}

// ---------------------------------------------------------------------------
// Kernel 1: pre0[t,b,:] = x[t,b,:] @ Wx0^T + bx0   (256000 x 80, K=440)
// Memory-bound (450 MB of x, read ONCE -> non-temporal loads so x does not
// evict pre0 from the 192 MB L2; pre0 stores keep default RT hint so the
// recurrence kernel finds them in L2).  f16 WMMA, f32 accumulation.
// 8 waves/block, each wave owns one 16-row M-tile and all 5 N-tiles.
// ---------------------------------------------------------------------------
__global__ __launch_bounds__(256) void pre0_gemm(
    const float* __restrict__ x, const _Float16* __restrict__ W16,
    const float* __restrict__ bx0, float* __restrict__ pre0)
{
    const int lane = threadIdx.x & 31;
    const int wave = threadIdx.x >> 5;
    const int m    = lane & 15;
    const int half = lane >> 4;
    const long tile = (long)blockIdx.x * 8 + wave;          // [0, 16000)
    const float* __restrict__ arow = x + (tile * 16 + m) * D_IN;

    v8f acc[5];
#pragma unroll
    for (int n = 0; n < 5; ++n) {
        const float b = bx0[n * 16 + m];                    // bias depends on N only
#pragma unroll
        for (int r = 0; r < 8; ++r) acc[n][r] = b;
    }

#pragma unroll 1
    for (int kc = 0; kc < 14; ++kc) {
        const int k0 = kc * 32;
        // A fragment, 16x32 f16: lane<16 -> K=0-15 ; lane>=16 -> K=16-31
        v16h a;
        if (kc < 13) {
#pragma unroll
            for (int e = 0; e < 8; ++e)
                a[e]     = (_Float16)__builtin_nontemporal_load(&arow[k0 + 16 * half + e]);
#pragma unroll
            for (int e = 0; e < 8; ++e)
                a[8 + e] = (_Float16)__builtin_nontemporal_load(&arow[k0 + 16 * half + 8 + e]);
        } else {                                            // K tail 416..439, pad to 447
#pragma unroll
            for (int e = 0; e < 16; ++e) {
                int k = k0 + 16 * half + e;
                a[e] = (k < D_IN) ? (_Float16)__builtin_nontemporal_load(&arow[k])
                                  : (_Float16)0.0f;
            }
        }
#pragma unroll
        for (int n = 0; n < 5; ++n) {
            // B fragment 32x16: lane -> N=lane%16, K=16*(lane/16)+e (contiguous)
            const v16h bfrag = *(const v16h*)&W16[(n * 16 + m) * KPAD + k0 + 16 * half];
            acc[n] = __builtin_amdgcn_wmma_f32_16x16x32_f16(
                false, a, false, bfrag, (short)0, acc[n], false, false);
        }
    }

    float* __restrict__ orow = pre0 + tile * 16 * G4;       // C/D: M = r + 8*half, N = n*16+m
#pragma unroll
    for (int n = 0; n < 5; ++n)
#pragma unroll
        for (int r = 0; r < 8; ++r)
            orow[(r + 8 * half) * G4 + n * 16 + m] = acc[n][r];
}

// ---------------------------------------------------------------------------
// Kernel 2: fused recurrence for BOTH layers.  16 independent workgroups
// (one wave each) own 16 batch rows apiece and walk all 1000 steps.
// Per step: 5 WMMAs (layer0 Uh0) + 10 WMMAs (layer1 Wx1 + Uh1), all weight
// B-fragments resident in VGPRs; LDS bounce for the gate transpose.
// The next step's 5 KB pre0 row is pulled ahead with global_prefetch
// (zero VGPR cost, no counter) so its latency hides under this step's work.
// ---------------------------------------------------------------------------
__global__ __launch_bounds__(32) void lstm_recur_fused(
    const float* __restrict__ pre0,
    const float* __restrict__ Uh0,
    const float* __restrict__ Wx1, const float* __restrict__ bx1,
    const float* __restrict__ Uh1,
    float* __restrict__ out)
{
    __shared__ alignas(32) float    gbuf[16 * G4];      // 16x80 gates (one layer at a time)
    __shared__ alignas(32) _Float16 hbuf[16 * 32];      // 16x20 h, zero-padded to K=32

    const int lane = threadIdx.x;
    const int m    = lane & 15;
    const int half = lane >> 4;
    const int b0   = blockIdx.x * 16;

    for (int i = lane; i < 16 * 32; i += 32) hbuf[i] = (_Float16)0.0f;
    __syncthreads();

    // Persistent B-fragments: B[k][n] = W[n_col][k], K=20 zero-padded to 32.
    v16h fU0[5], fW1[5], fU1[5];
#pragma unroll
    for (int n = 0; n < 5; ++n) {
#pragma unroll
        for (int e = 0; e < 16; ++e) {
            const int k   = 16 * half + e;
            const int col = n * 16 + m;
            _Float16 u0 = (_Float16)0.f, w1 = (_Float16)0.f, u1 = (_Float16)0.f;
            if (k < HID) {
                u0 = (_Float16)Uh0[col * HID + k];
                w1 = (_Float16)Wx1[col * HID + k];
                u1 = (_Float16)Uh1[col * HID + k];
            }
            fU0[n][e] = u0; fW1[n][e] = w1; fU1[n][e] = u1;
        }
    }
    float bias1[5];
#pragma unroll
    for (int n = 0; n < 5; ++n) bias1[n] = bx1[n * 16 + m];

    v16h hA0, hA1;                                      // h state as A-fragments
#pragma unroll
    for (int e = 0; e < 16; ++e) { hA0[e] = (_Float16)0.f; hA1[e] = (_Float16)0.f; }

    float c0[10], c1[10];                               // cell state, 10 (m,j) items/lane
#pragma unroll
    for (int q = 0; q < 10; ++q) { c0[q] = 0.f; c1[q] = 0.f; }

#pragma unroll 1
    for (int t = 0; t < T_STEPS; ++t) {
        const float* __restrict__ prow = pre0 + ((long)t * BATCH + b0) * G4;

        // ---- layer 0: gates = pre0 + h0 @ Uh0^T ----
        v8f acc[5];
#pragma unroll
        for (int n = 0; n < 5; ++n)
#pragma unroll
            for (int r = 0; r < 8; ++r)
                acc[n][r] = prow[(r + 8 * half) * G4 + n * 16 + m];

        // prefetch next step's contiguous 5120 B pre0 slab (40 cachelines)
        {
            const int  tn = (t + 1 < T_STEPS) ? (t + 1) : t;
            const char* pnext = (const char*)(pre0 + ((long)tn * BATCH + b0) * G4);
            __builtin_prefetch(pnext + lane * 128, 0, 3);
            __builtin_prefetch(pnext + 4096 + lane * 128, 0, 3);
        }

#pragma unroll
        for (int n = 0; n < 5; ++n)
            acc[n] = __builtin_amdgcn_wmma_f32_16x16x32_f16(
                false, hA0, false, fU0[n], (short)0, acc[n], false, false);

#pragma unroll
        for (int n = 0; n < 5; ++n)
#pragma unroll
            for (int r = 0; r < 8; ++r)
                gbuf[(r + 8 * half) * G4 + n * 16 + m] = acc[n][r];
        __syncthreads();

#pragma unroll
        for (int q = 0; q < 10; ++q) {                  // 320 items / 32 lanes
            const int item = lane + 32 * q;
            const int mi = item / HID;
            const int ji = item - mi * HID;
            const float gf = sigmoid_f(gbuf[mi * G4 +           ji]);
            const float gi = sigmoid_f(gbuf[mi * G4 + HID     + ji]);
            const float go = sigmoid_f(gbuf[mi * G4 + 2 * HID + ji]);
            const float gc = tanh_f   (gbuf[mi * G4 + 3 * HID + ji]);
            const float c  = gi * gc + gf * c0[q];
            c0[q] = c;
            hbuf[mi * 32 + ji] = (_Float16)(go * tanh_f(c));
        }
        __syncthreads();

#pragma unroll
        for (int e = 0; e < 16; ++e) {                  // rebuild hA0 (transpose via LDS)
            const int k = 16 * half + e;
            hA0[e] = hbuf[m * 32 + k];                  // pad cols already zero
        }
        __syncthreads();

        // ---- layer 1: gates = bx1 + h0_t @ Wx1^T + h1 @ Uh1^T ----
        v8f acc1[5];
#pragma unroll
        for (int n = 0; n < 5; ++n)
#pragma unroll
            for (int r = 0; r < 8; ++r) acc1[n][r] = bias1[n];
#pragma unroll
        for (int n = 0; n < 5; ++n)
            acc1[n] = __builtin_amdgcn_wmma_f32_16x16x32_f16(
                false, hA0, false, fW1[n], (short)0, acc1[n], false, false);
#pragma unroll
        for (int n = 0; n < 5; ++n)
            acc1[n] = __builtin_amdgcn_wmma_f32_16x16x32_f16(
                false, hA1, false, fU1[n], (short)0, acc1[n], false, false);

#pragma unroll
        for (int n = 0; n < 5; ++n)
#pragma unroll
            for (int r = 0; r < 8; ++r)
                gbuf[(r + 8 * half) * G4 + n * 16 + m] = acc1[n][r];
        __syncthreads();

        float* __restrict__ orow = out + (long)t * (BATCH * HID) + b0 * HID;
#pragma unroll
        for (int q = 0; q < 10; ++q) {
            const int item = lane + 32 * q;
            const int mi = item / HID;
            const int ji = item - mi * HID;
            const float gf = sigmoid_f(gbuf[mi * G4 +           ji]);
            const float gi = sigmoid_f(gbuf[mi * G4 + HID     + ji]);
            const float go = sigmoid_f(gbuf[mi * G4 + 2 * HID + ji]);
            const float gc = tanh_f   (gbuf[mi * G4 + 3 * HID + ji]);
            const float c  = gi * gc + gf * c1[q];
            c1[q] = c;
            const float h  = go * tanh_f(c);
            hbuf[mi * 32 + ji] = (_Float16)h;
            __builtin_nontemporal_store(h, &orow[item]);    // coalesced, NT (write-once)
        }
        __syncthreads();

#pragma unroll
        for (int e = 0; e < 16; ++e) {
            const int k = 16 * half + e;
            hA1[e] = hbuf[m * 32 + k];
        }
        __syncthreads();
    }
}

extern "C" void kernel_launch(void* const* d_in, const int* in_sizes, int n_in,
                              void* d_out, int out_size, void* d_ws, size_t ws_size,
                              hipStream_t stream) {
    const float* x   = (const float*)d_in[0];
    const float* Wx0 = (const float*)d_in[1];
    const float* bx0 = (const float*)d_in[2];
    const float* Uh0 = (const float*)d_in[3];
    const float* Wx1 = (const float*)d_in[4];
    const float* bx1 = (const float*)d_in[5];
    const float* Uh1 = (const float*)d_in[6];
    float* out = (float*)d_out;

    // workspace layout: pre0 (f32, 1000*256*80 = 81,920,000 B), then f16 Wx0 copy
    float*    pre0 = (float*)d_ws;
    _Float16* W16  = (_Float16*)((char*)d_ws + (size_t)T_STEPS * BATCH * G4 * sizeof(float));

    cvt_w0_f16<<<(G4 * KPAD + 255) / 256, 256, 0, stream>>>(Wx0, W16);
    pre0_gemm<<<(T_STEPS * BATCH / 16) / 8, 256, 0, stream>>>(x, W16, bx0, pre0);
    lstm_recur_fused<<<BATCH / 16, 32, 0, stream>>>(pre0, Uh0, Wx1, bx1, Uh1, out);
    (void)in_sizes; (void)n_in; (void)out_size; (void)ws_size;
}
